// BitNetLinear_1735166787590
// MI455X (gfx1250) — compile-verified, matrix operand order checked by reading, not run on hardware
//
#include <hip/hip_runtime.h>

typedef __attribute__((ext_vector_type(16))) __bf16 v16bf;
typedef __attribute__((ext_vector_type(8)))  __bf16 v8bf;
typedef __attribute__((ext_vector_type(8)))  float  v8f;

#define LAMBDA_VAL 0.5f

constexpr int Mdim = 8192;   // B * S
constexpr int Ndim = 4096;   // D_OUT
constexpr int Kdim = 4096;   // D_IN
constexpr int WN_ELEMS = Ndim * Kdim;   // 16777216
constexpr int XN_ELEMS = Mdim * Kdim;   // 33554432

// ---------------- workspace layout (bytes) ----------------
// 0        : float   sumAbs
// 256      : float   partials[2048]
// 16384    : ushort  eff_bf16[N*K]      (33,554,432 B)
// 16384+33554432 : ushort x_bf16[M*K]   (67,108,864 B)
constexpr size_t WS_PARTIALS = 256;
constexpr size_t WS_EFF      = 16384;
constexpr size_t WS_XBF      = WS_EFF + (size_t)WN_ELEMS * 2;

__device__ inline unsigned short f2bf(float f) {
  unsigned u = __float_as_uint(f);
  u = u + 0x7FFFu + ((u >> 16) & 1u);   // round-to-nearest-even
  return (unsigned short)(u >> 16);
}

// ---------------- 1) deterministic |W| reduction ----------------
__global__ void abssum_partial(const float4* __restrict__ w, float* __restrict__ partials) {
  float s = 0.f;
  const int stride = gridDim.x * blockDim.x;
  for (int i = blockIdx.x * blockDim.x + threadIdx.x; i < WN_ELEMS / 4; i += stride) {
    float4 v = w[i];
    s += fabsf(v.x) + fabsf(v.y) + fabsf(v.z) + fabsf(v.w);
  }
  for (int off = 16; off > 0; off >>= 1) s += __shfl_down(s, off, 32);
  __shared__ float red[8];
  if ((threadIdx.x & 31) == 0) red[threadIdx.x >> 5] = s;
  __syncthreads();
  if (threadIdx.x == 0) {
    float t = 0.f;
#pragma unroll
    for (int i = 0; i < 8; ++i) t += red[i];
    partials[blockIdx.x] = t;
  }
}

__global__ void abssum_final(const float* __restrict__ partials, float* __restrict__ sumAbs) {
  float s = 0.f;
  for (int i = threadIdx.x; i < 2048; i += 256) s += partials[i];
  for (int off = 16; off > 0; off >>= 1) s += __shfl_down(s, off, 32);
  __shared__ float red[8];
  if ((threadIdx.x & 31) == 0) red[threadIdx.x >> 5] = s;
  __syncthreads();
  if (threadIdx.x == 0) {
    float t = 0.f;
#pragma unroll
    for (int i = 0; i < 8; ++i) t += red[i];
    sumAbs[0] = t;
  }
}

// ---------------- 2) quantize + blend + alpha -> bf16 ----------------
__global__ void quant_convert(const float4* __restrict__ w,
                              const float* __restrict__ alpha,
                              const float* __restrict__ sumAbs,
                              uint2* __restrict__ eff) {
  const int i = blockIdx.x * blockDim.x + threadIdx.x;  // one float4
  const float thr = sumAbs[0] * (1.0f / (float)WN_ELEMS);
  const float a = alpha[0];
  float4 v = w[i];
  float r[4] = {v.x, v.y, v.z, v.w};
  unsigned short b[4];
#pragma unroll
  for (int j = 0; j < 4; ++j) {
    float wv = r[j];
    float q = (fabsf(wv) > thr) ? copysignf(1.0f, wv) : 0.0f;
    float e = ((1.0f - LAMBDA_VAL) * wv + LAMBDA_VAL * q) * a;
    b[j] = f2bf(e);
  }
  uint2 o;
  o.x = (unsigned)b[0] | ((unsigned)b[1] << 16);
  o.y = (unsigned)b[2] | ((unsigned)b[3] << 16);
  eff[i] = o;
}

// ---------------- 3) x fp32 -> bf16 ----------------
__global__ void x_convert(const float4* __restrict__ x, uint2* __restrict__ xb) {
  const int i = blockIdx.x * blockDim.x + threadIdx.x;
  float4 v = x[i];
  uint2 o;
  o.x = (unsigned)f2bf(v.x) | ((unsigned)f2bf(v.y) << 16);
  o.y = (unsigned)f2bf(v.z) | ((unsigned)f2bf(v.w) << 16);
  xb[i] = o;
}

// ---------------- 4) bf16 WMMA GEMM: out = x @ eff^T + bias ----------------
// Block tile 256x128, 8 waves in a 4(M)x2(N) grid, 64x64 per wave:
//   16 v_wmma_f32_16x16x32_bf16 per 16 ds_load_b128 (1.0 WMMA per LDS read).
// A/B tiles staged via GLOBAL_LOAD_ASYNC_TO_LDS_B128 (ASYNCcnt), double buffered.
constexpr int BM = 256, BN = 128, BK = 64;
constexpr int ROW_BYTES   = 144;              // 128 B payload (64 bf16) + 16 B pad
constexpr int TILE_A      = BM * ROW_BYTES;   // 36864
constexpr int TILE_B      = BN * ROW_BYTES;   // 18432
constexpr int BUF_BYTES   = TILE_A + TILE_B;  // 55296
// total LDS = 2 * BUF_BYTES = 110592 B  (< 320 KB per WGP)

__device__ inline v16bf load_frag(const char* p) {
  v8bf lo = *(const v8bf*)p;          // K = {0..7}   (lanes 0-15) / {8..15}  (lanes 16-31)
  v8bf hi = *(const v8bf*)(p + 32);   // K = {16..23} (lanes 0-15) / {24..31} (lanes 16-31)
  return __builtin_shufflevector(lo, hi, 0, 1, 2, 3, 4, 5, 6, 7,
                                 8, 9, 10, 11, 12, 13, 14, 15);
}

__global__ void __launch_bounds__(256)
gemm_bf16(const uint4* __restrict__ xbf,    // M x K bf16 (8 bf16 per uint4)
          const uint4* __restrict__ effbf,  // N x K bf16
          const float* __restrict__ bias,
          float* __restrict__ out) {
  __shared__ uint4 smem[2 * BUF_BYTES / 16];

  const int tid  = threadIdx.x;
  const int lane = tid & 31;
  const int wv   = tid >> 5;
  const int wm   = wv >> 1;  // 0..3 : 64-row band of M
  const int wn   = wv & 1;   // 0..1 : 64-col band of N
  const int m0   = blockIdx.y * BM;
  const int n0   = blockIdx.x * BN;

  // Loader: A tile = 256 rows x 128 B = 2048 16-B chunks (8/thread),
  //         B tile = 128 rows x 128 B = 1024 16-B chunks (4/thread).
  // chunk c = tid + 256*j -> row = (tid>>3) + 32*j, seg = tid&7 (affine in j).
  const int r0  = tid >> 3;
  const int seg = tid & 7;
  const unsigned smemBase = (unsigned)(uintptr_t)(void*)smem;  // flat low32 == LDS offset
  const unsigned offA0 = (unsigned)(m0 + r0) * (unsigned)(Kdim * 2) + (unsigned)seg * 16u;
  const unsigned offB0 = (unsigned)(n0 + r0) * (unsigned)(Kdim * 2) + (unsigned)seg * 16u;
  const unsigned ldsA0 = smemBase + (unsigned)(r0 * ROW_BYTES + seg * 16);
  const unsigned ldsB0 = smemBase + (unsigned)(TILE_A + r0 * ROW_BYTES + seg * 16);
  constexpr unsigned GSTRIDE = 32u * (unsigned)(Kdim * 2);  // 32 rows of global, per j
  constexpr unsigned LSTRIDE = 32u * (unsigned)ROW_BYTES;   // 32 rows of LDS, per j

  const unsigned long long baseA = (unsigned long long)(uintptr_t)xbf;
  const unsigned long long baseB = (unsigned long long)(uintptr_t)effbf;

  v8f acc[4][4];
#pragma unroll
  for (int i = 0; i < 4; ++i)
#pragma unroll
    for (int j = 0; j < 4; ++j) acc[i][j] = (v8f)(0.0f);

  // async-stage one K-step (buffer selected by bufByte, K window by kByte)
  auto issue_tile = [&](unsigned bufByte, unsigned kByte) {
#pragma unroll
    for (unsigned j = 0; j < 8; ++j) {
      asm volatile("global_load_async_to_lds_b128 %0, %1, %2"
                   :: "v"(ldsA0 + bufByte + j * LSTRIDE),
                      "v"(offA0 + kByte + j * GSTRIDE),
                      "s"(baseA)
                   : "memory");
    }
#pragma unroll
    for (unsigned j = 0; j < 4; ++j) {
      asm volatile("global_load_async_to_lds_b128 %0, %1, %2"
                   :: "v"(ldsB0 + bufByte + j * LSTRIDE),
                      "v"(offB0 + kByte + j * GSTRIDE),
                      "s"(baseB)
                   : "memory");
    }
  };

  // prologue: stage K-step 0 into buffer 0
  issue_tile(0u, 0u);
  asm volatile("s_wait_asynccnt 0x0" ::: "memory");
  __syncthreads();

  const int nK = Kdim / BK;  // 64 iterations
  // per-lane fragment offset inside a 16-row block:
  // lanes 0-15 -> K{0..7,16..23}, lanes 16-31 -> K{8..15,24..31} (16-bit A/B layout)
  const int fragOff = (lane & 15) * ROW_BYTES + (lane >> 4) * 16;

  int cur = 0;
#pragma unroll 1
  for (int ks = 0; ks < nK; ++ks) {
    const bool more = (ks + 1 < nK);
    if (more) issue_tile((unsigned)((cur ^ 1) * BUF_BYTES), (unsigned)((ks + 1) * (BK * 2)));

    const char* base = (const char*)smem + cur * BUF_BYTES;
#pragma unroll
    for (int s = 0; s < 2; ++s) {  // two K=32 sub-steps inside the 64-wide tile
      const char* aB = base + (wm * 64) * ROW_BYTES + s * 64 + fragOff;
      const char* bB = base + TILE_A + (wn * 64) * ROW_BYTES + s * 64 + fragOff;

      v16bf aF[4], bF[4];
#pragma unroll
      for (int mt = 0; mt < 4; ++mt) aF[mt] = load_frag(aB + mt * 16 * ROW_BYTES);
#pragma unroll
      for (int nt = 0; nt < 4; ++nt) bF[nt] = load_frag(bB + nt * 16 * ROW_BYTES);

#pragma unroll
      for (int mt = 0; mt < 4; ++mt)
#pragma unroll
        for (int nt = 0; nt < 4; ++nt)
          acc[mt][nt] = __builtin_amdgcn_wmma_f32_16x16x32_bf16(
              false, aF[mt], false, bF[nt], (short)0, acc[mt][nt], false, false);
    }

    if (more) asm volatile("s_wait_asynccnt 0x0" ::: "memory");
    __syncthreads();
    cur ^= 1;
  }

  // epilogue: C layout = VGPR v (M row), lane&15 (N col), lane>>4 selects M+8 band
  const int rowBase = m0 + wm * 64 + (lane >> 4) * 8;
  const int colBase = n0 + wn * 64 + (lane & 15);
#pragma unroll
  for (int nt = 0; nt < 4; ++nt) {
    const int col = colBase + nt * 16;
    const float bv = bias[col];
#pragma unroll
    for (int mt = 0; mt < 4; ++mt) {
      const int row = rowBase + mt * 16;
#pragma unroll
      for (int v = 0; v < 8; ++v) {
        out[(size_t)(row + v) * Ndim + col] = acc[mt][nt][v] + bv;
      }
    }
  }
}

// ---------------- launch ----------------
extern "C" void kernel_launch(void* const* d_in, const int* in_sizes, int n_in,
                              void* d_out, int out_size, void* d_ws, size_t ws_size,
                              hipStream_t stream) {
  const float* x      = (const float*)d_in[0];
  const float* wfp    = (const float*)d_in[1];
  const float* bias   = (const float*)d_in[2];
  const float* alpha  = (const float*)d_in[3];
  float* out          = (float*)d_out;
  char*  ws           = (char*)d_ws;

  float*          sumAbs   = (float*)ws;
  float*          partials = (float*)(ws + WS_PARTIALS);
  unsigned short* effbf    = (unsigned short*)(ws + WS_EFF);
  unsigned short* xbf      = (unsigned short*)(ws + WS_XBF);

  // 1) threshold reduction (deterministic two-stage)
  abssum_partial<<<2048, 256, 0, stream>>>((const float4*)wfp, partials);
  abssum_final<<<1, 256, 0, stream>>>(partials, sumAbs);

  // 2) eff = blend(quantize(W)) * alpha  -> bf16
  quant_convert<<<WN_ELEMS / 4 / 256, 256, 0, stream>>>(
      (const float4*)wfp, alpha, sumAbs, (uint2*)effbf);

  // 3) x -> bf16 (once; GEMM working set then fits in 192 MB L2)
  x_convert<<<XN_ELEMS / 4 / 256, 256, 0, stream>>>((const float4*)x, (uint2*)xbf);

  // 4) WMMA GEMM + bias
  dim3 grid(Ndim / BN, Mdim / BM);
  gemm_bf16<<<grid, 256, 0, stream>>>((const uint4*)xbf, (const uint4*)effbf, bias, out);
}
